// CustomAttn_57827439673381
// MI455X (gfx1250) — matrix operand order repaired and checked by reading, not compiled, not run on hardware
//
#include <hip/hip_runtime.h>

// ---------------------------------------------------------------------------
// CustomAttn for MI455X (gfx1250): bf16 WMMA pipeline, double-buffered LDS
//   qkv = x @ w_in^T          (WMMA GEMM, f32 accum, ping-pong LDS)
//   qk-RMSNorm(flat 2048) + RoPE + scatter to [B,NH,S,HD] bf16
//   flash attention (QK^T and P@V via WMMA, online softmax, causal)
//   out = attn @ w_out^T      (WMMA GEMM, f32 out)
// ---------------------------------------------------------------------------

typedef __attribute__((ext_vector_type(16))) __bf16 v16bf;
typedef __attribute__((ext_vector_type(8)))  __bf16 v8bf;
typedef __attribute__((ext_vector_type(8)))  float  v8f;

constexpr int kB = 2, kS = 2048, kHID = 2048, kNH = 16, kHD = 128;
constexpr float kLn10000 = 9.210340371976184f;

__device__ __forceinline__ __bf16 f2bf(float f) {
  unsigned u = __builtin_bit_cast(unsigned, f);
  u += 0x7FFFu + ((u >> 16) & 1u);            // round-to-nearest-even
  unsigned short h = (unsigned short)(u >> 16);
  return __builtin_bit_cast(__bf16, h);
}

// A fragment, 16x32 bf16, source row-major with leading dim `ld`.
// lanes 0-15 -> K {0..15}; lanes 16-31 -> K {16..31}
__device__ __forceinline__ v16bf load_a_frag(const __bf16* s, int ld) {
  const int lane = threadIdx.x & 31;
  const int m = lane & 15;
  const int kh = (lane >> 4) << 4;            // 0 or 16
  v16bf f;
#pragma unroll
  for (int j = 0; j < 8; ++j) {
    const int kb = kh + 2 * j;
    f[2 * j]     = s[m * ld + kb];
    f[2 * j + 1] = s[m * ld + kb + 1];
  }
  return f;
}

// B fragment, 32x16 bf16, element (k,n) = s[n*ld + k]  (source is N x K)
__device__ __forceinline__ v16bf load_b_frag_nk(const __bf16* s, int ld) {
  const int lane = threadIdx.x & 31;
  const int n = lane & 15;
  const int kh = (lane >> 4) << 4;            // 0 or 16
  v16bf f;
#pragma unroll
  for (int j = 0; j < 8; ++j) {
    f[2 * j]     = s[n * ld + kh + 2 * j];
    f[2 * j + 1] = s[n * ld + kh + 2 * j + 1];
  }
  return f;
}

// B fragment, 32x16 bf16, element (k,n) = s[k*ld + n]  (source is K x N)
__device__ __forceinline__ v16bf load_b_frag_kn(const __bf16* s, int ld) {
  const int lane = threadIdx.x & 31;
  const int n = lane & 15;
  const int kh = (lane >> 4) << 4;
  v16bf f;
#pragma unroll
  for (int j = 0; j < 8; ++j) {
    f[2 * j]     = s[(kh + 2 * j) * ld + n];
    f[2 * j + 1] = s[(kh + 2 * j + 1) * ld + n];
  }
  return f;
}

// ---------------------------------------------------------------------------
// f32 -> bf16 conversion, 8 elements / thread-iteration (b128 in, b128 out)
// n must be a multiple of 8 (true for all tensors here).
// ---------------------------------------------------------------------------
__global__ void cvt_f32_bf16(const float* __restrict__ src,
                             __bf16* __restrict__ dst, size_t n) {
  const size_t n8 = n >> 3;
  size_t i = (size_t)blockIdx.x * blockDim.x + threadIdx.x;
  const size_t stride = (size_t)gridDim.x * blockDim.x;
  for (; i < n8; i += stride) {
    const float4 a = *((const float4*)src + 2 * i);
    const float4 b = *((const float4*)src + 2 * i + 1);
    v8bf o = {f2bf(a.x), f2bf(a.y), f2bf(a.z), f2bf(a.w),
              f2bf(b.x), f2bf(b.y), f2bf(b.z), f2bf(b.w)};
    *((v8bf*)dst + i) = o;
  }
}

// ---------------------------------------------------------------------------
// C[M,N] f32 = A[M,K] bf16 (row-major) * B[N,K]^T bf16 (row-major NT GEMM)
// Block tile 128x128, 8 waves (4 along M x 2 along N), wave tile 32x64.
// K-step 64, ping-pong LDS double buffering: one barrier per K-step,
// global loads for step k+1 issued before the 16 WMMAs of step k,
// global_prefetch_b8 two steps ahead.  Dims are exact multiples of tiles.
// ---------------------------------------------------------------------------
constexpr int GT_M = 128, GT_N = 128, GT_K = 64, G_LDA = GT_K + 8;  // 72

__global__ void __launch_bounds__(256)
gemm_bf16_nt(const __bf16* __restrict__ A, const __bf16* __restrict__ Bm,
             float* __restrict__ C, int M, int N, int K) {
  __shared__ __attribute__((aligned(16))) __bf16 As[2][GT_M * G_LDA];
  __shared__ __attribute__((aligned(16))) __bf16 Bs[2][GT_M * G_LDA];
  const int t = threadIdx.x;
  const int wave = t >> 5;
  const int wm = wave & 3;
  const int wn = wave >> 2;
  const int mBase = blockIdx.y * GT_M;
  const int nBase = blockIdx.x * GT_N;

  const v8f vzero = {};
  v8f acc[2][4];
#pragma unroll
  for (int i = 0; i < 2; ++i)
#pragma unroll
    for (int j = 0; j < 4; ++j) acc[i][j] = vzero;

  v8bf areg[4], breg[4];
  auto fetch = [&](int k0) {
#pragma unroll
    for (int vv = 0; vv < 4; ++vv) {
      const int i = t * 32 + vv * 8;
      const int r = i >> 6, c = i & 63;
      areg[vv] = *(const v8bf*)&A[(size_t)(mBase + r) * K + k0 + c];
      breg[vv] = *(const v8bf*)&Bm[(size_t)(nBase + r) * K + k0 + c];
    }
  };
  auto stage = [&](int buf) {
#pragma unroll
    for (int vv = 0; vv < 4; ++vv) {
      const int i = t * 32 + vv * 8;
      const int r = i >> 6, c = i & 63;
      *(v8bf*)&As[buf][r * G_LDA + c] = areg[vv];
      *(v8bf*)&Bs[buf][r * G_LDA + c] = breg[vv];
    }
  };

  const int nK = K / GT_K;
  fetch(0);
  stage(0);
  __syncthreads();

  int cur = 0;
  for (int kt = 0; kt < nK; ++kt) {
    if (kt + 1 < nK) {
      fetch((kt + 1) * GT_K);                       // overlap with WMMAs below
      if (kt + 2 < nK) {                            // gfx1250 global_prefetch_b8
        const int r0 = (t * 32) >> 6;
        __builtin_prefetch(&A[(size_t)(mBase + r0) * K + (kt + 2) * GT_K], 0, 1);
        __builtin_prefetch(&Bm[(size_t)(nBase + r0) * K + (kt + 2) * GT_K], 0, 1);
      }
    }
#pragma unroll
    for (int ch = 0; ch < 2; ++ch) {                // two 32-deep WMMA chunks
      v16bf af[2], bfr[4];
#pragma unroll
      for (int mt = 0; mt < 2; ++mt)
        af[mt] = load_a_frag(&As[cur][(wm * 32 + mt * 16) * G_LDA + ch * 32], G_LDA);
#pragma unroll
      for (int nt = 0; nt < 4; ++nt)
        bfr[nt] = load_b_frag_nk(&Bs[cur][(wn * 64 + nt * 16) * G_LDA + ch * 32], G_LDA);
#pragma unroll
      for (int mt = 0; mt < 2; ++mt)
#pragma unroll
        for (int nt = 0; nt < 4; ++nt)
          acc[mt][nt] = __builtin_amdgcn_wmma_f32_16x16x32_bf16(
              false, af[mt], false, bfr[nt], (short)0, acc[mt][nt], false, false);
    }
    if (kt + 1 < nK) stage(cur ^ 1);
    __syncthreads();
    cur ^= 1;
  }

  // epilogue: one base pointer per output row, stores at immediate offsets
  const int lane = t & 31;
  const int half = lane >> 4;
  const int col0 = nBase + wn * 64 + (lane & 15);
#pragma unroll
  for (int mt = 0; mt < 2; ++mt)
#pragma unroll
    for (int vg = 0; vg < 8; ++vg) {
      const int row = mBase + wm * 32 + mt * 16 + vg + 8 * half;
      float* p = C + (size_t)row * N + col0;
#pragma unroll
      for (int nt = 0; nt < 4; ++nt) p[nt * 16] = acc[mt][nt][vg];
    }
}

// ---------------------------------------------------------------------------
// qk RMSNorm over flat 2048 + RoPE (rotate-half) + scatter to [B,NH,S,HD]
// One block per (b,s) token row; 256 threads.
// ---------------------------------------------------------------------------
__global__ void __launch_bounds__(256)
qknorm_rope(const float* __restrict__ qkv, const float* __restrict__ qw,
            const float* __restrict__ kw, __bf16* __restrict__ qo,
            __bf16* __restrict__ ko, __bf16* __restrict__ vo) {
  const int bs = blockIdx.x;                   // b*S + s
  const int b = bs / kS;
  const int s = bs % kS;
  const int t = threadIdx.x;
  const float* qr = qkv + (size_t)bs * (3 * kHID);
  const float* kr = qr + kHID;
  const float* vr = qr + 2 * kHID;

  __shared__ float red[256];
  float sq = 0.f, sk = 0.f;
#pragma unroll
  for (int j = 0; j < 8; ++j) {
    const int e = t + j * 256;
    const float a = qr[e]; sq += a * a;
    const float c = kr[e]; sk += c * c;
  }
  red[t] = sq; __syncthreads();
  for (int ofs = 128; ofs > 0; ofs >>= 1) {
    if (t < ofs) red[t] += red[t + ofs];
    __syncthreads();
  }
  const float ssq = red[0]; __syncthreads();
  red[t] = sk; __syncthreads();
  for (int ofs = 128; ofs > 0; ofs >>= 1) {
    if (t < ofs) red[t] += red[t + ofs];
    __syncthreads();
  }
  const float ssk = red[0];

  const float rq = rsqrtf(ssq / (float)kHID + 1e-5f);
  const float rk = rsqrtf(ssk / (float)kHID + 1e-5f);
  const float pos = (float)s;

#pragma unroll
  for (int j = 0; j < 8; ++j) {
    const int e = t + j * 256;
    const int h = e >> 7;                      // head
    const int d = e & 127;                     // dim in head
    const int d0 = d & 63;
    const int ep = (d < 64) ? (e + 64) : (e - 64);
    const float freq = __expf(-(float)d0 * (kLn10000 / 64.f));
    const float ang = pos * freq;
    const float cs = cosf(ang), sn = sinf(ang);

    const float qn  = qr[e]  * rq * qw[e];
    const float qpn = qr[ep] * rq * qw[ep];
    const float kn  = kr[e]  * rk * kw[e];
    const float kpn = kr[ep] * rk * kw[ep];
    const float qro = (d < 64) ? (qn * cs - qpn * sn) : (qn * cs + qpn * sn);
    const float kro = (d < 64) ? (kn * cs - kpn * sn) : (kn * cs + kpn * sn);

    const size_t dst = ((size_t)(b * kNH + h) * kS + s) * kHD + d;
    qo[dst] = f2bf(qro);
    ko[dst] = f2bf(kro);
    vo[dst] = f2bf(vr[e]);
  }
}

// ---------------------------------------------------------------------------
// Flash attention, causal.  One block = (b,h) x 64 query rows, 4 waves,
// each wave owns 16 query rows.  K/V tiles of 32 keys, ping-pong LDS
// double buffering -> one barrier per tile; global fetch for tile kt+1
// overlaps the WMMAs of tile kt.  Uniform trip count per block.
// ---------------------------------------------------------------------------
constexpr int QT = 64, KT = 32, F_LD = kHD + 8;  // 136

__global__ void __launch_bounds__(128)
flash_attn(const __bf16* __restrict__ q, const __bf16* __restrict__ k,
           const __bf16* __restrict__ v, __bf16* __restrict__ o) {
  __shared__ __attribute__((aligned(16))) __bf16 Qs[QT * F_LD];
  __shared__ __attribute__((aligned(16))) __bf16 Ks[2][KT * F_LD];
  __shared__ __attribute__((aligned(16))) __bf16 Vs[2][KT * F_LD];
  __shared__ __attribute__((aligned(16))) __bf16 Ps[4 * 16 * KT];

  const int qtile = blockIdx.x;
  const int bh = blockIdx.y;                   // b*NH + h
  const int b = bh >> 4;
  const int h = bh & 15;
  const int qbase = qtile * QT;
  const int t = threadIdx.x;
  const int wave = t >> 5;
  const int lane = t & 31;
  const int half = lane >> 4;
  const int nn = lane & 15;

  const size_t head_off = (size_t)bh * kS * kHD;
  const __bf16* qp = q + head_off;
  const __bf16* kp = k + head_off;
  const __bf16* vp = v + head_off;

  // stage the 64x128 Q tile (vectorized)
#pragma unroll
  for (int vv = 0; vv < 8; ++vv) {
    const int i = t * 64 + vv * 8;
    const int r = i >> 7, c = i & 127;
    *(v8bf*)&Qs[r * F_LD + c] = *(const v8bf*)&qp[(size_t)(qbase + r) * kHD + c];
  }

  v8bf kreg[4], vreg[4];
  auto fetchKV = [&](int kb) {
#pragma unroll
    for (int vv = 0; vv < 4; ++vv) {
      const int i = t * 32 + vv * 8;
      const int r = i >> 7, c = i & 127;
      kreg[vv] = *(const v8bf*)&kp[(size_t)(kb + r) * kHD + c];
      vreg[vv] = *(const v8bf*)&vp[(size_t)(kb + r) * kHD + c];
    }
  };
  auto stageKV = [&](int buf) {
#pragma unroll
    for (int vv = 0; vv < 4; ++vv) {
      const int i = t * 32 + vv * 8;
      const int r = i >> 7, c = i & 127;
      *(v8bf*)&Ks[buf][r * F_LD + c] = kreg[vv];
      *(v8bf*)&Vs[buf][r * F_LD + c] = vreg[vv];
    }
  };

  fetchKV(0);
  stageKV(0);
  __syncthreads();

  v16bf qa[4];
#pragma unroll
  for (int c = 0; c < 4; ++c)
    qa[c] = load_a_frag(&Qs[(wave * 16) * F_LD + c * 32], F_LD);

  const v8f vzero = {};
  v8f oacc[8];
  float mrow[8], lrow[8];
#pragma unroll
  for (int i = 0; i < 8; ++i) { oacc[i] = vzero; mrow[i] = -1e30f; lrow[i] = 0.f; }

  const float scale = 0.08838834764831845f;    // 1/sqrt(128)
  const int nkt = (qbase + QT) / KT;           // causal: keys up to qbase+63
  __bf16* Pw = &Ps[wave * 16 * KT];

  int cur = 0;
  for (int kt = 0; kt < nkt; ++kt) {
    const int kb = kt * KT;
    if (kt + 1 < nkt) {
      fetchKV(kb + KT);                        // overlap with compute below
      if (kt + 2 < nkt) {
        const int r0 = (t * 32) >> 7;
        __builtin_prefetch(&kp[(size_t)(kb + 2 * KT + r0) * kHD], 0, 1);
        __builtin_prefetch(&vp[(size_t)(kb + 2 * KT + r0) * kHD], 0, 1);
      }
    }

    // scores: S(16x32) = Q(16x128) K^T, as two 16x16 WMMA tiles
    v8f sfr[2];
#pragma unroll
    for (int kn = 0; kn < 2; ++kn) {
      v8f accs = vzero;
#pragma unroll
      for (int c = 0; c < 4; ++c) {
        v16bf bfr = load_b_frag_nk(&Ks[cur][(kn * 16) * F_LD + c * 32], F_LD);
        accs = __builtin_amdgcn_wmma_f32_16x16x32_bf16(
            false, qa[c], false, bfr, (short)0, accs, false, false);
      }
      sfr[kn] = accs;
    }

    // scale + causal mask + online softmax (row = C-frag row of this lane)
#pragma unroll
    for (int vg = 0; vg < 8; ++vg) {
      const int row = qbase + wave * 16 + vg + 8 * half;
      float s0 = sfr[0][vg] * scale;
      float s1 = sfr[1][vg] * scale;
      if (kb + nn > row)      s0 = -1e30f;
      if (kb + 16 + nn > row) s1 = -1e30f;
      float mx = fmaxf(s0, s1);
#pragma unroll
      for (int d = 1; d < 16; d <<= 1) mx = fmaxf(mx, __shfl_xor(mx, d, 32));
      const float mnew = fmaxf(mrow[vg], mx);
      const float alpha = __expf(mrow[vg] - mnew);
      const float p0 = __expf(s0 - mnew);
      const float p1 = __expf(s1 - mnew);
      float rs = p0 + p1;
#pragma unroll
      for (int d = 1; d < 16; d <<= 1) rs += __shfl_xor(rs, d, 32);
      lrow[vg] = lrow[vg] * alpha + rs;
      mrow[vg] = mnew;
#pragma unroll
      for (int nd = 0; nd < 8; ++nd) oacc[nd][vg] *= alpha;
      const int prow = vg + 8 * half;
      Pw[prow * KT + nn]      = f2bf(p0);      // re-layout P via LDS
      Pw[prow * KT + 16 + nn] = f2bf(p1);
    }

    // O += P(16x32) @ V(32x128); wave-private LDS RAW is in-order (DScnt)
    const v16bf pa = load_a_frag(Pw, KT);
#pragma unroll
    for (int nd = 0; nd < 8; ++nd) {
      v16bf vb = load_b_frag_kn(&Vs[cur][nd * 16], F_LD);
      oacc[nd] = __builtin_amdgcn_wmma_f32_16x16x32_bf16(
          false, pa, false, vb, (short)0, oacc[nd], false, false);
    }

    if (kt + 1 < nkt) stageKV(cur ^ 1);
    __syncthreads();
    cur ^= 1;
  }

  // epilogue: O /= l, write [B,S,NH*HD] bf16, immediate-offset stores
#pragma unroll
  for (int vg = 0; vg < 8; ++vg) {
    const int row = qbase + wave * 16 + vg + 8 * half;
    const float inv = 1.f / lrow[vg];
    __bf16* pdst = o + (size_t)(b * kS + row) * (kNH * kHD) + h * kHD + nn;
#pragma unroll
    for (int nd = 0; nd < 8; ++nd) pdst[nd * 16] = f2bf(oacc[nd][vg] * inv);
  }
}

// ---------------------------------------------------------------------------
extern "C" void kernel_launch(void* const* d_in, const int* in_sizes, int n_in,
                              void* d_out, int out_size, void* d_ws, size_t ws_size,
                              hipStream_t stream) {
  const float* x     = (const float*)d_in[0];
  const float* w_in  = (const float*)d_in[1];
  const float* w_out = (const float*)d_in[2];
  const float* qw    = (const float*)d_in[3];
  const float* kw    = (const float*)d_in[4];
  float* out = (float*)d_out;
  (void)in_sizes; (void)n_in; (void)out_size; (void)ws_size;

  const size_t xe    = (size_t)kB * kS * kHID;         // 8388608
  const size_t wine  = (size_t)3 * kNH * kHD * kHID;   // 12582912
  const size_t woute = (size_t)kHID * kNH * kHD;       // 4194304
  const size_t qkve  = (size_t)kB * kS * 3 * kNH * kHD;
  const size_t he    = (size_t)kB * kNH * kS * kHD;    // 8388608
  const size_t ae    = (size_t)kB * kS * kNH * kHD;

  char* ws = (char*)d_ws;
  size_t off = 0;
  auto take = [&](size_t bytes) -> char* {
    char* p = ws + off;
    off = (off + bytes + 255) & ~(size_t)255;
    return p;
  };
  __bf16* x_bf    = (__bf16*)take(xe * 2);
  __bf16* win_bf  = (__bf16*)take(wine * 2);
  __bf16* wout_bf = (__bf16*)take(woute * 2);
  float*  qkv     = (float*)take(qkve * 4);
  __bf16* q_bf    = (__bf16*)take(he * 2);
  __bf16* k_bf    = (__bf16*)take(he * 2);
  __bf16* v_bf    = (__bf16*)take(he * 2);
  __bf16* a_bf    = (__bf16*)take(ae * 2);

  cvt_f32_bf16<<<dim3(1024), 256, 0, stream>>>(x, x_bf, xe);
  cvt_f32_bf16<<<dim3(1024), 256, 0, stream>>>(w_in, win_bf, wine);
  cvt_f32_bf16<<<dim3(1024), 256, 0, stream>>>(w_out, wout_bf, woute);

  // qkv[4096, 6144] = x[4096, 2048] @ w_in[6144, 2048]^T
  gemm_bf16_nt<<<dim3(6144 / GT_N, 4096 / GT_M), 256, 0, stream>>>(
      x_bf, win_bf, qkv, 4096, 6144, 2048);

  qknorm_rope<<<dim3(kB * kS), 256, 0, stream>>>(qkv, qw, kw, q_bf, k_bf, v_bf);

  flash_attn<<<dim3(kS / QT, kB * kNH), 128, 0, stream>>>(q_bf, k_bf, v_bf, a_bf);

  // out[4096, 2048] = attn[4096, 2048] @ w_out[2048, 2048]^T
  gemm_bf16_nt<<<dim3(2048 / GT_N, 4096 / GT_M), 256, 0, stream>>>(
      a_bf, wout_bf, out, 4096, 2048, 2048);
}
